// VSS_Block_57088705298650
// MI455X (gfx1250) — matrix operand order repaired and checked by reading, not compiled, not run on hardware
//
#include <hip/hip_runtime.h>
#include <math.h>

typedef _Float16 h16;
typedef __attribute__((ext_vector_type(16))) _Float16 v16h;
typedef __attribute__((ext_vector_type(8)))  _Float16 v8h;
typedef __attribute__((ext_vector_type(8)))  float    v8f;
typedef __attribute__((ext_vector_type(4)))  float    f4;

#define B_   8
#define C_   96
#define H_   64
#define W_   64
#define L_   4096
#define D_   192
#define NST  16
#define KD   4
#define NPAD 48           // x_proj N=38 padded to 48 for unguarded B-fragment loads
#define ROWS (B_*L_)      // 32768
#define EPSF 1e-5f

__device__ __forceinline__ float wsum32(float v){
  v += __shfl_xor(v, 1, 32);
  v += __shfl_xor(v, 2, 32);
  v += __shfl_xor(v, 4, 32);
  v += __shfl_xor(v, 8, 32);
  v += __shfl_xor(v, 16, 32);
  return v;
}
__device__ __forceinline__ float siluf(float x){ return x / (1.f + expf(-x)); }
__device__ __forceinline__ int trp(int l){ return ((l & 63) << 6) | (l >> 6); } // (l%64)*64 + l/64

// ---------------- f32 -> f16 convert ----------------
__global__ void cvt_f16(const float* __restrict__ a, h16* __restrict__ o, int n){
  int i = blockIdx.x * 256 + threadIdx.x;
  if (i < n) o[i] = (h16)a[i];
}

// ---------------- x_proj_w [4,38,192] -> f16 padded [4,48,192] ----------------
__global__ void cvt_xpw_pad(const float* __restrict__ a, h16* __restrict__ o){
  int i = blockIdx.x * 256 + threadIdx.x;
  if (i >= KD*NPAD*D_) return;
  int d = i % D_, c = (i / D_) % NPAD, k = i / (D_*NPAD);
  o[i] = (c < 38) ? (h16)a[(k*38 + c)*D_ + d] : (h16)0.f;
}

// ---------------- LayerNorm over C=96, f16 out (wave per row) ----------------
__global__ __launch_bounds__(256) void ln96_h(const float* __restrict__ x, const float* __restrict__ g,
                                              const float* __restrict__ b, h16* __restrict__ o, int rows){
  int wv = threadIdx.x >> 5, lane = threadIdx.x & 31;
  int row = blockIdx.x * 8 + wv;
  if (row >= rows) return;
  const float* xr = x + (long)row * C_;
  float v0 = xr[lane], v1 = xr[lane+32], v2 = xr[lane+64];
  float mu = wsum32(v0 + v1 + v2) * (1.f/96.f);
  float d0 = v0-mu, d1 = v1-mu, d2 = v2-mu;
  float var = wsum32(d0*d0 + d1*d1 + d2*d2) * (1.f/96.f);
  float inv = rsqrtf(var + EPSF);
  h16* orow = o + (long)row * C_;
  orow[lane]    = (h16)(d0*inv*g[lane]    + b[lane]);
  orow[lane+32] = (h16)(d1*inv*g[lane+32] + b[lane+32]);
  orow[lane+64] = (h16)(d2*inv*g[lane+64] + b[lane+64]);
}

// ---------------- LayerNorm over C=96, f32 out ----------------
__global__ __launch_bounds__(256) void ln96_f(const float* __restrict__ x, const float* __restrict__ g,
                                              const float* __restrict__ b, float* __restrict__ o, int rows){
  int wv = threadIdx.x >> 5, lane = threadIdx.x & 31;
  int row = blockIdx.x * 8 + wv;
  if (row >= rows) return;
  const float* xr = x + (long)row * C_;
  float v0 = xr[lane], v1 = xr[lane+32], v2 = xr[lane+64];
  float mu = wsum32(v0 + v1 + v2) * (1.f/96.f);
  float d0 = v0-mu, d1 = v1-mu, d2 = v2-mu;
  float var = wsum32(d0*d0 + d1*d1 + d2*d2) * (1.f/96.f);
  float inv = rsqrtf(var + EPSF);
  float* orow = o + (long)row * C_;
  orow[lane]    = d0*inv*g[lane]    + b[lane];
  orow[lane+32] = d1*inv*g[lane+32] + b[lane+32];
  orow[lane+64] = d2*inv*g[lane+64] + b[lane+64];
}

// ---------------- out_norm (over D=192) fused with *silu(z), f16 out ----------------
__global__ __launch_bounds__(256) void onorm_silu_h(const float* __restrict__ ym, const float* __restrict__ xz,
                                                    const float* __restrict__ g, const float* __restrict__ b,
                                                    h16* __restrict__ o, int rows){
  int wv = threadIdx.x >> 5, lane = threadIdx.x & 31;
  int row = blockIdx.x * 8 + wv;
  if (row >= rows) return;
  const float* xr = ym + (long)row * D_;
  float v[6], s = 0.f;
#pragma unroll
  for (int i = 0; i < 6; ++i){ v[i] = xr[lane + 32*i]; s += v[i]; }
  float mu = wsum32(s) * (1.f/192.f);
  float var = 0.f;
#pragma unroll
  for (int i = 0; i < 6; ++i){ v[i] -= mu; var += v[i]*v[i]; }
  var = wsum32(var) * (1.f/192.f);
  float inv = rsqrtf(var + EPSF);
#pragma unroll
  for (int i = 0; i < 6; ++i){
    int c = lane + 32*i;
    float z = xz[(long)row*(2*D_) + D_ + c];
    float yv = v[i]*inv*g[c] + b[c];
    o[(long)row*D_ + c] = (h16)(yv * siluf(z));
  }
}

// ---------------- WMMA GEMM: C[m,n] = sum_k A[m,k]*B[n,k], k contiguous in both ----------------
// f16 operands / f32 accum. One 16x16 tile per wave, 4 waves per block, full K unroll.
// Requirements: M % 16 == 0; B has >= ceil16(N) rows allocated (loads unguarded, stores guarded);
// A,B rows 16B-aligned (am, bn multiples of 8; base 256B-aligned).
template<int KDIM>
__global__ __launch_bounds__(128) void wmma_gemm_f16(
    const h16* __restrict__ A, const h16* __restrict__ Bm, float* __restrict__ Cm,
    int M, int N,
    long am, long bn, long cm, long cn,
    long aBo, long aBi, long bBo, long bBi, long cBo, long cBi, int innerMod)
{
  int z  = blockIdx.z;
  int zo = z / innerMod, zi = z - zo * innerMod;
  A  += zo*aBo + zi*aBi;
  Bm += zo*bBo + zi*bBi;
  Cm += zo*cBo + zi*cBi;
  int wv = threadIdx.x >> 5;
  int lane = threadIdx.x & 31;
  int r = lane & 15, hl = lane >> 4;
  int m0 = (blockIdx.y * 4 + wv) * 16;
  int n0 = blockIdx.x * 16;
  if (m0 >= M) return;                    // wave-uniform
  const h16* Arow = A  + (long)(m0 + r) * am;
  const h16* Brow = Bm + (long)(n0 + r) * bn;
  v8f acc = {};
#pragma unroll
  for (int k0 = 0; k0 < KDIM; k0 += 32) {
    // A 16x32 f16 layout: lanes 0-15 hold K {0..7,16..23}; lanes 16-31 hold K {8..15,24..31}
    v8h alo = *(const v8h*)(Arow + k0 + 8*hl);
    v8h ahi = *(const v8h*)(Arow + k0 + 16 + 8*hl);
    // B 32x16 f16 layout: lanes 0-15 hold K {0..7,16..23}; lanes 16-31 hold K {8..15,24..31} (N = lane&15)
    v8h blo = *(const v8h*)(Brow + k0 + 8*hl);
    v8h bhi = *(const v8h*)(Brow + k0 + 16 + 8*hl);
    v16h a = __builtin_shufflevector(alo, ahi, 0,1,2,3,4,5,6,7,8,9,10,11,12,13,14,15);
    v16h b = __builtin_shufflevector(blo, bhi, 0,1,2,3,4,5,6,7,8,9,10,11,12,13,14,15);
    acc = __builtin_amdgcn_wmma_f32_16x16x32_f16(false, a, false, b, (short)0, acc, false, false);
  }
#pragma unroll
  for (int j = 0; j < 8; ++j) {
    int m = m0 + j + hl*8, n = n0 + r;
    if (n < N) Cm[(long)m*cm + (long)n*cn] = acc[j];
  }
}

// ---------------- depthwise 3x3 conv + SiLU; in: xz[row,0:192]; out: xi[b,192,L] ----------------
__global__ void dwconv_silu(const float* __restrict__ xz, const float* __restrict__ w,
                            const float* __restrict__ bias, float* __restrict__ xi){
  long gid = (long)blockIdx.x * 256 + threadIdx.x;
  if (gid >= (long)B_*D_*L_) return;
  int l = gid & (L_-1);
  int d = (gid >> 12) % D_;
  int b = gid / ((long)D_*L_);
  int h = l >> 6, wc = l & 63;
  float s = bias[d];
#pragma unroll
  for (int dy = -1; dy <= 1; ++dy){
    int hh = h + dy;
    if ((unsigned)hh >= 64u) continue;
#pragma unroll
    for (int dx = -1; dx <= 1; ++dx){
      int ww = wc + dx;
      if ((unsigned)ww >= 64u) continue;
      s += w[d*9 + (dy+1)*3 + (dx+1)] * xz[((long)b*L_ + hh*64 + ww)*(2*D_) + d];
    }
  }
  xi[gid] = siluf(s);
}

// ---------------- build xs (4 scan directions), transposed [b,k,l,192] f16, d-coalesced stores ----
__global__ void build_xs(const float* __restrict__ xi, h16* __restrict__ xs){
  long gid = (long)blockIdx.x * 256 + threadIdx.x;
  if (gid >= (long)B_*D_*L_) return;
  int d = gid % D_;
  int l = (gid / D_) % L_;
  int b = gid / ((long)D_*L_);
  const float* xr = xi + ((long)b*D_ + d) * L_;
  h16* xo = xs + ((long)b*KD*L_ + l) * D_ + d;
  long ks = (long)L_ * D_;
  int rl = L_-1-l;
  xo[0]     = (h16)xr[l];
  xo[ks]    = (h16)xr[trp(l)];
  xo[2*ks]  = (h16)xr[rl];
  xo[3*ks]  = (h16)xr[trp(rl)];
}

// ---------------- selective scan: lane per (b,k,d,n); vectorized streams + prefetch ----------------
__global__ __launch_bounds__(256) void scan_kernel(
    const float* __restrict__ xdbl,   // [B,4,38,L]
    const float* __restrict__ xi,     // [B,192,L]
    const float* __restrict__ dtw,    // [4,192,6]
    const float* __restrict__ dtb,    // [4,192]
    const float* __restrict__ Alogs,  // [4*192,16]
    const float* __restrict__ Ds,     // [4*192]
    float* __restrict__ y4)           // [B,4,192,L]
{
  long gid = (long)blockIdx.x * 256 + threadIdx.x;   // B*4*192*16 lanes
  int n  = gid & 15;
  long t = gid >> 4;
  int d  = t % D_;
  int bz = t / D_;          // b*4+k
  int b  = bz >> 2, k = bz & 3;
  const float* xd  = xdbl + (long)bz * 38 * L_;
  const float* xin = xi   + ((long)b*D_ + d) * L_;
  float* yo        = y4   + ((long)bz*D_ + d) * L_;
  int kd = k*D_ + d;
  float A = -expf(Alogs[kd*16 + n]);
  float w0 = dtw[kd*6+0], w1 = dtw[kd*6+1], w2 = dtw[kd*6+2];
  float w3 = dtw[kd*6+3], w4 = dtw[kd*6+4], w5 = dtw[kd*6+5];
  float bias = dtb[kd], Dv = Ds[kd];
  const float* Bp = xd + (6 + n) * L_;
  const float* Cp = xd + (22 + n) * L_;
  float h = 0.f;
  for (int l4 = 0; l4 < L_; l4 += 4){
    if ((l4 & 127) == 0 && l4 + 512 < L_){
      __builtin_prefetch(Bp + l4 + 512, 0, 1);      // -> global_prefetch_b8
      __builtin_prefetch(Cp + l4 + 512, 0, 1);
      __builtin_prefetch(xd + l4 + 512, 0, 1);
    }
    f4 bv = *(const f4*)(Bp + l4);
    f4 cv = *(const f4*)(Cp + l4);
    f4 fv[6];
#pragma unroll
    for (int r2 = 0; r2 < 6; ++r2) fv[r2] = *(const f4*)(xd + r2*L_ + l4);
    f4 yv;
#pragma unroll
    for (int q = 0; q < 4; ++q){
      int l  = l4 + q;
      int j  = (k >= 2) ? (L_-1-l) : l;
      int lp = (k & 1) ? trp(j) : j;
      float u = xin[lp];
      float f = bias + w0*fv[0][q] + w1*fv[1][q] + w2*fv[2][q]
                     + w3*fv[3][q] + w4*fv[4][q] + w5*fv[5][q];
      float dt = (f > 20.f) ? f : log1pf(expf(f));   // softplus
      h = expf(dt * A) * h + (dt * u) * bv[q];
      float p = h * cv[q];
      p += __shfl_xor(p, 1, 32);
      p += __shfl_xor(p, 2, 32);
      p += __shfl_xor(p, 4, 32);
      p += __shfl_xor(p, 8, 32);
      yv[q] = p + Dv * u;
    }
    if (n == 0) *(f4*)(yo + l4) = yv;
  }
}

// ---------------- merge 4 directions -> ym[b,l,d] ----------------
__global__ void merge_kernel(const float* __restrict__ y4, float* __restrict__ ym){
  long gid = (long)blockIdx.x * 256 + threadIdx.x;
  if (gid >= (long)B_*D_*L_) return;
  int l = gid & (L_-1);
  int d = (gid >> 12) % D_;
  int b = gid / ((long)D_*L_);
  const float* base = y4 + ((long)b*KD*D_ + d) * L_;
  long ks = (long)D_ * L_;
  int tl = trp(l);
  float v = base[l] + base[2*ks + (L_-1-l)] + base[ks + tl] + base[3*ks + (L_-1-tl)];
  ym[((long)b*L_ + l)*D_ + d] = v;
}

// ---------------- x1 = inp*skip + attn ----------------
__global__ void x1_kernel(const float* __restrict__ inp, const float* __restrict__ ss,
                          const float* __restrict__ attn, float* __restrict__ x1){
  long i = (long)blockIdx.x * 256 + threadIdx.x;
  if (i >= (long)ROWS*C_) return;
  int c = i % C_;
  x1[i] = inp[i] * ss[c] + attn[i];
}

// ---------------- CAB conv1: [b,l,96] -> gelu -> [b,32,L] ----------------
__global__ void cab_conv1(const float* __restrict__ x2n, const float* __restrict__ w,
                          const float* __restrict__ bias, float* __restrict__ cb1){
  long gid = (long)blockIdx.x * 256 + threadIdx.x;
  if (gid >= (long)B_*32*L_) return;
  int l = gid & (L_-1);
  int o = (gid >> 12) & 31;
  int b = gid / (32L*L_);
  int h = l >> 6, wc = l & 63;
  float s = bias[o];
  for (int dy = -1; dy <= 1; ++dy){
    int hh = h + dy;
    if ((unsigned)hh >= 64u) continue;
    for (int dx = -1; dx <= 1; ++dx){
      int ww = wc + dx;
      if ((unsigned)ww >= 64u) continue;
      const float* xp = x2n + ((long)b*L_ + hh*64 + ww) * C_;
      const float* wp = w + (long)o*C_*9 + (dy+1)*3 + (dx+1);
      float acc = 0.f;
      for (int c = 0; c < C_; ++c) acc += xp[c] * wp[(long)c*9];
      s += acc;
    }
  }
  cb1[gid] = 0.5f * s * (1.f + erff(s * 0.70710678118f));   // exact GELU
}

// ---------------- CAB conv2: [b,32,L] -> [b,96,L] ----------------
__global__ void cab_conv2(const float* __restrict__ cb1, const float* __restrict__ w,
                          const float* __restrict__ bias, float* __restrict__ cb2){
  long gid = (long)blockIdx.x * 256 + threadIdx.x;
  if (gid >= (long)B_*C_*L_) return;
  int l = gid & (L_-1);
  int o = (gid >> 12) % C_;
  int b = gid / ((long)C_*L_);
  int h = l >> 6, wc = l & 63;
  float s = bias[o];
  for (int dy = -1; dy <= 1; ++dy){
    int hh = h + dy;
    if ((unsigned)hh >= 64u) continue;
    for (int dx = -1; dx <= 1; ++dx){
      int ww = wc + dx;
      if ((unsigned)ww >= 64u) continue;
      const float* xp = cb1 + (long)b*32*L_ + hh*64 + ww;
      const float* wp = w + (long)o*32*9 + (dy+1)*3 + (dx+1);
      float acc = 0.f;
      for (int c = 0; c < 32; ++c) acc += xp[(long)c*L_] * wp[(long)c*9];
      s += acc;
    }
  }
  cb2[gid] = s;
}

// ---------------- adaptive avg pool over HW ----------------
__global__ __launch_bounds__(256) void pool_kernel(const float* __restrict__ cb2, float* __restrict__ pooled){
  int bc = blockIdx.x;                    // B*C
  __shared__ float sm[256];
  float s = 0.f;
  for (int i = threadIdx.x; i < L_; i += 256) s += cb2[(long)bc*L_ + i];
  sm[threadIdx.x] = s;
  __syncthreads();
  for (int st = 128; st > 0; st >>= 1){
    if (threadIdx.x < st) sm[threadIdx.x] += sm[threadIdx.x + st];
    __syncthreads();
  }
  if (threadIdx.x == 0) pooled[bc] = sm[0] * (1.f/(float)L_);
}

// ---------------- channel attention (squeeze=30 -> 3 mid units) ----------------
__global__ void ca_kernel(const float* __restrict__ pooled,
                          const float* __restrict__ w1, const float* __restrict__ b1,
                          const float* __restrict__ w2, const float* __restrict__ b2,
                          float* __restrict__ cas){
  int b = blockIdx.x, c = threadIdx.x;    // 96 threads
  __shared__ float sp[C_], st[3];
  sp[c] = pooled[b*C_ + c];
  __syncthreads();
  if (c < 3){
    float s = b1[c];
    for (int j = 0; j < C_; ++j) s += w1[c*C_ + j] * sp[j];
    st[c] = fmaxf(s, 0.f);
  }
  __syncthreads();
  float s = b2[c];
  for (int j = 0; j < 3; ++j) s += w2[c*3 + j] * st[j];
  cas[b*C_ + c] = 1.f / (1.f + expf(-s));
}

// ---------------- final: out[b,c,h,w] = x1*skip2 + cb2*ca ----------------
__global__ void final_kernel(const float* __restrict__ x1, const float* __restrict__ ss2,
                             const float* __restrict__ cb2, const float* __restrict__ cas,
                             float* __restrict__ out){
  long i = (long)blockIdx.x * 256 + threadIdx.x;
  if (i >= (long)B_*C_*L_) return;
  int l = i & (L_-1);
  int c = (i >> 12) % C_;
  int b = i / ((long)C_*L_);
  out[i] = x1[((long)b*L_ + l)*C_ + c] * ss2[c] + cb2[i] * cas[b*C_ + c];
}

extern "C" void kernel_launch(void* const* d_in, const int* in_sizes, int n_in,
                              void* d_out, int out_size, void* d_ws, size_t ws_size,
                              hipStream_t stream) {
  (void)in_sizes; (void)n_in; (void)out_size; (void)ws_size;
  const float* x     = (const float*)d_in[0];   // [B,C,H,W] == raw [B,H,W,C] view
  const float* ln1g  = (const float*)d_in[1];
  const float* ln1b  = (const float*)d_in[2];
  const float* skip  = (const float*)d_in[3];
  const float* ln2g  = (const float*)d_in[4];
  const float* ln2b  = (const float*)d_in[5];
  const float* skip2 = (const float*)d_in[6];
  const float* ipw   = (const float*)d_in[7];   // [384,96]
  const float* cvw   = (const float*)d_in[8];   // [192,1,3,3]
  const float* cvb   = (const float*)d_in[9];
  const float* xpw   = (const float*)d_in[10];  // [4,38,192]
  const float* dtw   = (const float*)d_in[11];  // [4,192,6]
  const float* dtb   = (const float*)d_in[12];  // [4,192]
  const float* Alog  = (const float*)d_in[13];  // [768,16]
  const float* Ds    = (const float*)d_in[14];  // [768]
  const float* ong   = (const float*)d_in[15];
  const float* onb   = (const float*)d_in[16];
  const float* opw   = (const float*)d_in[17];  // [96,192]
  const float* cw1   = (const float*)d_in[18];  // [32,96,3,3]
  const float* cbi1  = (const float*)d_in[19];
  const float* cw2   = (const float*)d_in[20];  // [96,32,3,3]
  const float* cbi2  = (const float*)d_in[21];
  const float* caw1  = (const float*)d_in[22];  // [3,96,1,1]
  const float* cab1b = (const float*)d_in[23];
  const float* caw2  = (const float*)d_in[24];  // [96,3,1,1]
  const float* cab2b = (const float*)d_in[25];
  float* out = (float*)d_out;

  char* p = (char*)d_ws;
  auto alloc = [&](size_t bytes) -> void* {
    void* r = (void*)p;
    p += (bytes + 255) & ~(size_t)255;
    return r;
  };
  h16*   xn_h  = (h16*)  alloc((size_t)ROWS*C_*2);
  h16*   ipw_h = (h16*)  alloc((size_t)384*96*2);
  h16*   xpw_h = (h16*)  alloc((size_t)KD*NPAD*D_*2);       // zero-padded rows 38..47
  h16*   opw_h = (h16*)  alloc((size_t)96*192*2);
  float* xz    = (float*)alloc((size_t)ROWS*384*4);
  float* xiC   = (float*)alloc((size_t)B_*D_*L_*4);
  h16*   xs_t  = (h16*)  alloc((size_t)B_*KD*L_*D_*2);      // [b,k,l,192]
  float* xdbl  = (float*)alloc((size_t)B_*KD*38*L_*4);
  float* y4    = (float*)alloc((size_t)B_*KD*D_*L_*4);
  float* ym    = (float*)alloc((size_t)B_*L_*D_*4);
  h16*   y2h   = (h16*)  alloc((size_t)ROWS*D_*2);
  float* attn  = (float*)alloc((size_t)ROWS*C_*4);
  float* x1    = (float*)alloc((size_t)ROWS*C_*4);
  float* x2n   = (float*)alloc((size_t)ROWS*C_*4);
  float* cb1   = (float*)alloc((size_t)B_*32*L_*4);
  float* cb2   = (float*)alloc((size_t)B_*C_*L_*4);
  float* pooled= (float*)alloc((size_t)B_*C_*4);
  float* cas   = (float*)alloc((size_t)B_*C_*4);

  // weights -> f16 (x_proj padded)
  cvt_f16<<<(384*96+255)/256, 256, 0, stream>>>(ipw, ipw_h, 384*96);
  cvt_xpw_pad<<<(KD*NPAD*D_+255)/256, 256, 0, stream>>>(xpw, xpw_h);
  cvt_f16<<<(96*192+255)/256, 256, 0, stream>>>(opw, opw_h, 96*192);

  // ln1 -> xn (f16)
  ln96_h<<<ROWS/8, 256, 0, stream>>>(x, ln1g, ln1b, xn_h, ROWS);

  // in_proj: xz[ROWS,384] = xn[ROWS,96] @ ipw[384,96]^T   (WMMA, K=96)
  wmma_gemm_f16<96><<<dim3(384/16, ROWS/64, 1), 128, 0, stream>>>(
      xn_h, ipw_h, xz, ROWS, 384,
      /*am*/96, /*bn*/96, /*cm*/384, /*cn*/1,
      0, 0, 0, 0, 0, 0, 1);

  // depthwise conv + silu -> xiC[b,192,L]
  long ndw = (long)B_*D_*L_;
  dwconv_silu<<<(ndw+255)/256, 256, 0, stream>>>(xz, cvw, cvb, xiC);

  // 4-direction xs, transposed [b,k,l,192] (f16)
  build_xs<<<(ndw+255)/256, 256, 0, stream>>>(xiC, xs_t);

  // x_proj per (b,k): xdbl[c,l] = sum_d xs_t[l,d] * xpw[k][c,d]   (batched WMMA, K=192)
  wmma_gemm_f16<192><<<dim3(NPAD/16, L_/64, B_*KD), 128, 0, stream>>>(
      xs_t, xpw_h, xdbl, L_, 38,
      /*am*/D_, /*bn*/D_, /*cm*/1, /*cn*/L_,
      /*aBo*/(long)KD*L_*D_, /*aBi*/(long)L_*D_,
      /*bBo*/0,              /*bBi*/(long)NPAD*D_,
      /*cBo*/(long)KD*38*L_, /*cBi*/(long)38*L_, KD);

  // selective scan
  scan_kernel<<<(B_*KD*D_*NST)/256, 256, 0, stream>>>(xdbl, xiC, dtw, dtb, Alog, Ds, y4);

  // merge directions -> ym[b,l,192]
  merge_kernel<<<(ndw+255)/256, 256, 0, stream>>>(y4, ym);

  // out_norm * silu(z) -> y2 (f16)
  onorm_silu_h<<<ROWS/8, 256, 0, stream>>>(ym, xz, ong, onb, y2h, ROWS);

  // out_proj: attn[ROWS,96] = y2[ROWS,192] @ opw[96,192]^T   (WMMA, K=192)
  wmma_gemm_f16<192><<<dim3(96/16, ROWS/64, 1), 128, 0, stream>>>(
      y2h, opw_h, attn, ROWS, 96,
      /*am*/192, /*bn*/192, /*cm*/96, /*cn*/1,
      0, 0, 0, 0, 0, 0, 1);

  // residual 1
  long nres = (long)ROWS*C_;
  x1_kernel<<<(nres+255)/256, 256, 0, stream>>>(x, skip, attn, x1);

  // ln2 -> x2n (f32)
  ln96_f<<<ROWS/8, 256, 0, stream>>>(x1, ln2g, ln2b, x2n, ROWS);

  // CAB
  long nc1 = (long)B_*32*L_;
  cab_conv1<<<(nc1+255)/256, 256, 0, stream>>>(x2n, cw1, cbi1, cb1);
  cab_conv2<<<(nres+255)/256, 256, 0, stream>>>(cb1, cw2, cbi2, cb2);
  pool_kernel<<<B_*C_, 256, 0, stream>>>(cb2, pooled);
  ca_kernel<<<B_, C_, 0, stream>>>(pooled, caw1, cab1b, caw2, cab2b, cas);

  // final residual + layout transpose to [B,C,H,W]
  final_kernel<<<(nres+255)/256, 256, 0, stream>>>(x1, skip2, cb2, cas, out);
}
